// IntegralTransform_78950088835569
// MI455X (gfx1250) — matrix operand order repaired and checked by reading, not compiled, not run on hardware
//
#include <hip/hip_runtime.h>

#define KNBR 32
#define D_FEAT 64
#define D_HID 128
#define K1PAD 96     // 70 padded to 3 * 32 (column order: f0..f63, y0..2, x0..2, pad)
#define NEXT 80      // 64 out cols + 1 score col, padded to 5 * 16

#define EDGES_PER_BLOCK 128
#define QUERIES_PER_BLOCK 4
#define THREADS 256

// ws layout (u16 elements):
//   [0, 12288)            W1t permuted bf16 [128 n][96 k']
//   [12288, 22528)        W2t extended bf16 [80 n][128 k]
//   [22528, 22528+NIN*64) f_y bf16 [N_IN][64]
#define WS_W1T_U16   0
#define WS_W2T_U16   12288
#define WS_FY_U16    22528

typedef __attribute__((ext_vector_type(16))) __bf16 v16bf;
typedef __attribute__((ext_vector_type(8)))  float  v8f;
typedef __attribute__((ext_vector_type(4)))  unsigned int u32x4;

__device__ __forceinline__ unsigned short f2bf(float f) {
  unsigned int u = __float_as_uint(f);
  unsigned int r = 0x7FFFu + ((u >> 16) & 1u);   // round-to-nearest-even
  return (unsigned short)((u + r) >> 16);
}

// gfx1250 transcendental tanh (single trans op; v_nop covers the TRANS RAW hazard)
__device__ __forceinline__ float fast_tanh(float x) {
  float r;
  asm("v_tanh_f32 %0, %1\n\tv_nop" : "=v"(r) : "v"(x));
  return r;
}

__device__ __forceinline__ float gelu_tanh(float x) {
  const float kC = 0.7978845608028654f;          // sqrt(2/pi)
  float x3 = x * x * x;
  return 0.5f * x * (1.0f + fast_tanh(kC * (x + 0.044715f * x3)));
}

__device__ __forceinline__ unsigned lds_off(const void* p) {
  return (unsigned)(uintptr_t)p;                 // LDS aperture: low 32 bits = LDS addr
}

__device__ __forceinline__ void async_b128(unsigned lds_addr, const void* gptr) {
  unsigned long long ga = (unsigned long long)(uintptr_t)gptr;
  asm volatile("global_load_async_to_lds_b128 %0, %1, off"
               :: "v"(lds_addr), "v"(ga) : "memory");
}

union FragU { v16bf v; u32x4 u[2]; };

// A fragment (16x32 bf16, MxK) from row-major [row][k] LDS, stride in u16 elems.
__device__ __forceinline__ v16bf load_a(const unsigned short* base, int stride,
                                        int mrow0, int lane, int ks) {
  int row = mrow0 + (lane & 15);
  int kb = ks * 32 + ((lane >> 4) << 3);
  const unsigned short* p = base + row * stride + kb;
  FragU f;
  f.u[0] = *(const u32x4*)(p);
  f.u[1] = *(const u32x4*)(p + 16);
  return f.v;
}

// B fragment (32x16 bf16, KxN) from transposed row-major [n][k] LDS.
__device__ __forceinline__ v16bf load_b(const unsigned short* base, int stride,
                                        int ncol0, int lane, int ks) {
  int n = ncol0 + (lane & 15);
  int kb = ks * 32 + ((lane >> 4) << 3);
  const unsigned short* p = base + n * stride + kb;
  FragU f;
  f.u[0] = *(const u32x4*)(p);
  f.u[1] = *(const u32x4*)(p + 16);
  return f.v;
}

// ---------------- prep: one-time f32 -> bf16 tables in workspace ----------------
__global__ __launch_bounds__(THREADS)
void it_prep_kernel(const float* __restrict__ W1, const float* __restrict__ W2,
                    const float* __restrict__ av, const float* __restrict__ fy,
                    unsigned short* __restrict__ ws, int n_fy) {
  int g = blockIdx.x * THREADS + threadIdx.x;
  int nthreads = gridDim.x * THREADS;
  if (g < 12288) {                       // W1^T with column permutation f|y|x|pad
    int n = g / K1PAD, kp = g % K1PAD;
    float v;
    if (kp < 64)      v = W1[(kp + 6) * D_HID + n];   // feature rows
    else if (kp < 70) v = W1[(kp - 64) * D_HID + n];  // y (64..66), x (67..69)
    else              v = 0.0f;                       // pad
    ws[WS_W1T_U16 + g] = f2bf(v);
  } else if (g < 22528) {                // [W2 | a | 0]^T
    int i = g - 12288;
    int n = i / D_HID, k = i % D_HID;
    float v = (n < 64) ? W2[k * 64 + n] : ((n == 64) ? av[k] : 0.0f);
    ws[WS_W2T_U16 + (i)] = f2bf(v);
  }
  for (int i = g; i < n_fy; i += nthreads)
    ws[WS_FY_U16 + i] = f2bf(fy[i]);
}

// ---------------- main fused kernel ----------------
__global__ __launch_bounds__(THREADS)
void IntegralTransform_78950088835569_kernel(
    const float* __restrict__ y, const float* __restrict__ x,
    const float* __restrict__ b1, const float* __restrict__ b2,
    const int* __restrict__ nbr, const unsigned short* __restrict__ ws,
    float* __restrict__ out) {
  // LDS layout (bytes):
  //  [0,24576)       sEdge u16 [128][96]  (cols: f0..63, y0..2, x0..2, pad)
  //  [24576,69632)   sW1t [128][96] ++ sW2t [80][128]  (one contiguous async copy)
  //  [0,32768)       sM f32 [128][64]     (alias, after GEMM1 barrier)
  //  [32768,33280)   sScore f32 [128]     (alias)
  //  [33280,33792)   sAlpha f32 [128]     (alias)
  //  [69632,102400)  sH u16 [128][128]
  //  [102400,102912) sB1 f32 [128]
  //  [102912,103232) sB2e f32 [80]
  __shared__ __align__(16) char smem[103232];
  unsigned short* sEdge  = (unsigned short*)(smem);
  unsigned short* sW1t   = (unsigned short*)(smem + 24576);
  unsigned short* sW2t   = (unsigned short*)(smem + 49152);
  float*          sM     = (float*)(smem);
  float*          sScore = (float*)(smem + 32768);
  float*          sAlpha = (float*)(smem + 33280);
  unsigned short* sH     = (unsigned short*)(smem + 69632);
  float*          sB1    = (float*)(smem + 102400);
  float*          sB2e   = (float*)(smem + 102912);

  const int tid  = threadIdx.x;
  const int lane = tid & 31;
  const int wave = tid >> 5;
  const int blk  = blockIdx.x;
  const int e0   = blk * EDGES_PER_BLOCK;

  // ---- Stage 0a: async copy of bf16 weights (45056 B) into LDS ----
  {
    unsigned ldsW = lds_off(sW1t);
    const char* gW = (const char*)(ws + WS_W1T_U16);
    #pragma unroll
    for (int it = 0; it < 11; ++it) {              // 2816 x 16B chunks / 256 threads
      int c = tid + it * THREADS;
      async_b128(ldsW + c * 16, gW + c * 16);
    }
  }
  // ---- Stage 0b: async gather of bf16 neighbor features (128 B per edge) ----
  {
    const char* gF = (const char*)(ws + WS_FY_U16);
    unsigned ldsE = lds_off(sEdge);
    int sub = lane & 7;                            // 8 lanes per edge
    #pragma unroll
    for (int i = 0; i < 4; ++i) {                  // 4 edges per wave-instruction
      int el = wave * 16 + (lane >> 3) + i * 4;
      int idx = nbr[e0 + el];
      async_b128(ldsE + el * (K1PAD * 2) + sub * 16,
                 gF + (size_t)idx * (D_FEAT * 2) + sub * 16);
    }
  }
  // ---- Stage 0c: y/x/pad columns + biases (plain stores, disjoint from async) ----
  {
    int e = tid & 127;
    int part = tid >> 7;
    int eg = e0 + e;
    unsigned short* row = sEdge + e * K1PAD;
    if (part == 0) {
      int idx = nbr[eg];
      int qg = eg >> 5;
      #pragma unroll
      for (int j = 0; j < 3; ++j) row[64 + j] = f2bf(y[idx * 3 + j]);
      #pragma unroll
      for (int j = 0; j < 3; ++j) row[67 + j] = f2bf(x[qg * 3 + j]);
      #pragma unroll
      for (int c = 70; c < 80; ++c) row[c] = 0;
    } else {
      u32x4 z = {};
      *(u32x4*)(row + 80) = z;
      *(u32x4*)(row + 88) = z;
    }
  }
  if (tid < D_HID) sB1[tid] = b1[tid];
  if (tid < NEXT)  sB2e[tid] = (tid < 64) ? b2[tid] : 0.0f;
  asm volatile("s_wait_asynccnt 0x0" ::: "memory");
  __syncthreads();

  // ---- Stage 1: h = gelu(edge_in @ W1 + b1), per-wave 16x128 tile ----
  {
    v8f acc[8] = {};
    #pragma unroll
    for (int ks = 0; ks < 3; ++ks) {
      v16bf a = load_a(sEdge, K1PAD, wave * 16, lane, ks);
      #pragma unroll
      for (int nw = 0; nw < 8; ++nw) {
        v16bf b = load_b(sW1t, K1PAD, nw * 16, lane, ks);
        acc[nw] = __builtin_amdgcn_wmma_f32_16x16x32_bf16(
            false, a, false, b, (short)0, acc[nw], false, false);
      }
    }
    int mbase = wave * 16 + ((lane >> 4) << 3);
    int nloc  = lane & 15;
    #pragma unroll
    for (int nw = 0; nw < 8; ++nw) {
      int n = nw * 16 + nloc;
      float bias = sB1[n];
      #pragma unroll
      for (int r = 0; r < 8; ++r) {
        float v = acc[nw][r] + bias;
        sH[(mbase + r) * D_HID + n] = f2bf(gelu_tanh(v));
      }
    }
  }
  __syncthreads();

  // ---- Stage 2: [m | score] = h @ [W2 | a] + [b2 | 0] ----
  {
    v8f acc[5] = {};
    #pragma unroll
    for (int ks = 0; ks < 4; ++ks) {
      v16bf a = load_a(sH, D_HID, wave * 16, lane, ks);
      #pragma unroll
      for (int nw = 0; nw < 5; ++nw) {
        v16bf b = load_b(sW2t, D_HID, nw * 16, lane, ks);
        acc[nw] = __builtin_amdgcn_wmma_f32_16x16x32_bf16(
            false, a, false, b, (short)0, acc[nw], false, false);
      }
    }
    int mbase = wave * 16 + ((lane >> 4) << 3);
    int nloc  = lane & 15;
    #pragma unroll
    for (int nw = 0; nw < 5; ++nw) {
      int n = nw * 16 + nloc;
      float bias = sB2e[n];
      #pragma unroll
      for (int r = 0; r < 8; ++r) {
        float v = acc[nw][r] + bias;
        int m = mbase + r;
        if (n < 64)       sM[m * 64 + n] = v;   // aliases dead sEdge/sW1t (post-barrier)
        else if (n == 64) sScore[m] = v;
      }
    }
  }
  __syncthreads();

  // ---- Stage 3: softmax over each neighborhood of K=32 (one wave per query) ----
  if (tid < EDGES_PER_BLOCK) {
    float s = sScore[tid];
    float mx = s;
    #pragma unroll
    for (int off = 16; off > 0; off >>= 1) mx = fmaxf(mx, __shfl_xor(mx, off, 32));
    float e = __expf(s - mx);
    float sum = e;
    #pragma unroll
    for (int off = 16; off > 0; off >>= 1) sum += __shfl_xor(sum, off, 32);
    sAlpha[tid] = e / sum;
  }
  __syncthreads();

  // ---- Stage 4: out[q,n] = sum_k alpha[q,k] * m[q,k,n] ----
  {
    int q = tid >> 6;
    int n = tid & 63;
    const float* mrow = sM + (q * KNBR) * 64 + n;
    const float* al   = sAlpha + q * KNBR;
    float acc = 0.0f;
    #pragma unroll
    for (int k = 0; k < KNBR; ++k) acc += al[k] * mrow[k * 64];
    out[(blk * QUERIES_PER_BLOCK + q) * 64 + n] = acc;
  }
}

extern "C" void kernel_launch(void* const* d_in, const int* in_sizes, int n_in,
                              void* d_out, int out_size, void* d_ws, size_t ws_size,
                              hipStream_t stream) {
  const float* y   = (const float*)d_in[0];
  const float* x   = (const float*)d_in[1];
  const float* fy  = (const float*)d_in[2];
  const float* W1  = (const float*)d_in[3];
  const float* b1  = (const float*)d_in[4];
  const float* W2  = (const float*)d_in[5];
  const float* b2  = (const float*)d_in[6];
  const float* av  = (const float*)d_in[7];
  const int*   nbr = (const int*)d_in[8];
  float* out = (float*)d_out;
  unsigned short* ws = (unsigned short*)d_ws;

  const int n_fy    = in_sizes[2];                 // N_IN * 64
  const int n_edges = in_sizes[8];                 // N_OUT * K
  const int n_blocks = n_edges / EDGES_PER_BLOCK;  // 6250 for the reference sizes

  it_prep_kernel<<<512, THREADS, 0, stream>>>(W1, W2, av, fy, ws, n_fy);
  IntegralTransform_78950088835569_kernel<<<n_blocks, THREADS, 0, stream>>>(
      y, x, b1, b2, nbr, ws, out);
}
